// AdaptiveTokenFilter_83288005804243
// MI455X (gfx1250) — matrix operand order repaired and checked by reading, not compiled, not run on hardware
//
#include <hip/hip_runtime.h>
#include <hip/hip_bf16.h>

// ---------------------------------------------------------------------------
// AdaptiveTokenFilter for MI455X (gfx1250, wave32, WMMA)
//   B=8, S=4096, D=1024, H=2048, MAX_K=64, TAU=1, K_TAU=1
// Dominant cost: relu(emb@W1+b1)@W2 fused scorer -> v_wmma_f32_16x16x32_bf16
// N-tiled 4-wide: one A fragment feeds 4 WMMAs (4 H-column tiles) to cut
// LDS operand traffic per WMMA 4x and expose independent accumulator ILP.
// ---------------------------------------------------------------------------

#define B_DIM 8
#define S_DIM 4096
#define D_DIM 1024
#define H_DIM 2048
#define M_TOTAL (B_DIM * S_DIM)      // 32768 tokens
#define MAX_K 64

#define ROWS_PER_BLOCK 128           // 8 waves * 16 rows
#define SCORER_THREADS 256
#define NTILE 4                      // H-column tiles per wave iteration (64 cols)

typedef unsigned short ushort_t;
typedef __attribute__((ext_vector_type(16))) __bf16 v16bf;
typedef __attribute__((ext_vector_type(8)))  float  v8f;

// ---------------- helpers ----------------

__device__ __forceinline__ ushort_t f2bf(float f) {
    unsigned u = __float_as_uint(f);
    unsigned r = u + 0x7FFFu + ((u >> 16) & 1u);   // round-to-nearest-even
    return (ushort_t)(r >> 16);
}

__device__ __forceinline__ unsigned pcg_hash(unsigned x) {
    unsigned s = x * 747796405u + 2891336453u;
    unsigned w = ((s >> ((s >> 28u) + 4u)) ^ s) * 277803737u;
    return (w >> 22u) ^ w;
}

__device__ __forceinline__ float gumbel_from(unsigned h) {
    float u = (float)(h >> 8) * (1.0f / 16777216.0f);
    u = fminf(fmaxf(u, 1e-8f), 1.0f - 1e-8f);
    return -__logf(-__logf(u));
}

union AFrag { uint4 q[2]; v16bf v; };

// ---------------- kernel 1: W1 [D,H] f32 -> w1T [H,D] bf16 ----------------

__global__ void atf_convert_w1(const float* __restrict__ W1,
                               ushort_t* __restrict__ w1T) {
    int idx = blockIdx.x * blockDim.x + threadIdx.x;   // over D*H
    if (idx >= D_DIM * H_DIM) return;
    int d = idx / H_DIM;
    int h = idx - d * H_DIM;
    w1T[(size_t)h * D_DIM + d] = f2bf(W1[idx]);        // coalesced read
}

// ---------------- kernel 2: fused scorer (WMMA) ----------------
// logits[m] = sum_h relu(emb[m,:]@W1[:,h] + b1[h]) * W2[h] + b2

__global__ void atf_scorer(const float* __restrict__ emb,     // [M, D] f32
                           const ushort_t* __restrict__ w1T,  // [H, D] bf16
                           const float* __restrict__ b1,      // [H]
                           const float* __restrict__ w2,      // [H]
                           const float* __restrict__ b2,      // [1]
                           float* __restrict__ logits) {      // [M]
    extern __shared__ ushort_t aTile[];                 // [128][1024] bf16 = 256 KB
    const int tid  = threadIdx.x;
    const int wave = tid >> 5;
    const int lane = tid & 31;
    const int l15  = lane & 15;
    const int halfSel = lane >> 4;                      // 0 | 1
    const int mBase = blockIdx.x * ROWS_PER_BLOCK;

    // Stage & convert 128 rows of emb to bf16 in LDS (full-K resident:
    // CDNA5's 320KB-per-WGP LDS makes a 256KB tile viable).
    for (int i = tid * 4; i < ROWS_PER_BLOCK * D_DIM; i += SCORER_THREADS * 4) {
        const float4 f = *reinterpret_cast<const float4*>(emb + (size_t)mBase * D_DIM + i);
        unsigned lo = (unsigned)f2bf(f.x) | ((unsigned)f2bf(f.y) << 16);
        unsigned hi = (unsigned)f2bf(f.z) | ((unsigned)f2bf(f.w) << 16);
        *reinterpret_cast<uint2*>(&aTile[i]) = make_uint2(lo, hi);
    }
    __syncthreads();

    const int rowTile = wave * 16;                      // this wave's 16 rows
    float acc[8];
#pragma unroll
    for (int r = 0; r < 8; ++r) acc[r] = 0.0f;

    for (int hb = 0; hb < H_DIM; hb += 16 * NTILE) {
        const int col0 = hb + l15;                      // N index, tile 0
        // prefetch next B block into cache hierarchy (global_prefetch_b8)
        if (hb + 16 * NTILE < H_DIM)
            __builtin_prefetch(w1T + (size_t)(col0 + 16 * NTILE) * D_DIM, 0, 1);

        // Per-lane base pointers for the 4 B columns (K-contiguous rows of w1T).
        const ushort_t* bp[NTILE];
#pragma unroll
        for (int j = 0; j < NTILE; ++j)
            bp[j] = w1T + (size_t)(col0 + 16 * j) * D_DIM + halfSel * 16;

        v8f c[NTILE];
#pragma unroll
        for (int j = 0; j < NTILE; ++j) c[j] = {};

#pragma unroll 4
        for (int kb = 0; kb < D_DIM; kb += 32) {
            // A fragment (K order matches B): lanes0-15 hold K0..15,
            // lanes16-31 hold K16..31, row M = l15. One fragment feeds
            // all NTILE WMMAs (4x less LDS traffic per matrix op).
            AFrag a;
            const ushort_t* arow = &aTile[(rowTile + l15) * D_DIM + kb];
            a.q[0] = *reinterpret_cast<const uint4*>(arow + halfSel * 16);
            a.q[1] = *reinterpret_cast<const uint4*>(arow + halfSel * 16 + 8);
#pragma unroll
            for (int j = 0; j < NTILE; ++j) {
                // B fragment: lanes0-15 K0..15, lanes16-31 K16..31 of column j.
                const v16bf bfrag = *reinterpret_cast<const v16bf*>(bp[j] + kb);
                c[j] = __builtin_amdgcn_wmma_f32_16x16x32_bf16(
                           false, a.v, false, bfrag, (short)0, c[j], false, false);
            }
        }
        // epilogue: +b1, relu, *W2, accumulate per-row partial dot
#pragma unroll
        for (int j = 0; j < NTILE; ++j) {
            const int col = col0 + 16 * j;
            const float bias = b1[col];
            const float w2v  = w2[col];
#pragma unroll
            for (int r = 0; r < 8; ++r) {
                float v = c[j][r] + bias;               // C layout: row = r + halfSel*8
                v = v > 0.0f ? v : 0.0f;
                acc[r] = fmaf(v, w2v, acc[r]);
            }
        }
    }

    // Reduce over the 16 lanes holding each row's N-columns (xor<16 stays in half)
#pragma unroll
    for (int r = 0; r < 8; ++r) {
        float v = acc[r];
        v += __shfl_xor(v, 1, 32);
        v += __shfl_xor(v, 2, 32);
        v += __shfl_xor(v, 4, 32);
        v += __shfl_xor(v, 8, 32);
        acc[r] = v;
    }
    const float bb = b2[0];
    if (l15 == 0) {                                     // lanes 0 and 16
#pragma unroll
        for (int r = 0; r < 8; ++r)
            logits[mBase + rowTile + halfSel * 8 + r] = acc[r] + bb;
    }
}

// ---------------- kernel 3: learnable-k selection ----------------

__global__ void atf_kselect(const float* __restrict__ k_logits,
                            float* __restrict__ expected_k_out,
                            int* __restrict__ k_sel_out) {
    __shared__ float sh[MAX_K];
    int t = threadIdx.x;
    if (t < MAX_K) {
        float g = gumbel_from(pcg_hash(0x9E3779B9u + t));
        sh[t] = k_logits[t] + g;                        // K_TAU = 1
    }
    __syncthreads();
    if (t == 0) {
        float mx = -3.4e38f; int arg = 0;
        for (int i = 0; i < MAX_K; ++i)
            if (sh[i] > mx) { mx = sh[i]; arg = i; }
        float sum = 0.0f;
        for (int i = 0; i < MAX_K; ++i) sum += __expf(sh[i] - mx);
        float ek = 0.0f;
        for (int i = 0; i < MAX_K; ++i)
            ek += (__expf(sh[i] - mx) / sum) * (float)(i + 1);
        expected_k_out[0] = ek;
        k_sel_out[0] = arg + 1;
    }
}

// ---------------- kernel 4: per-batch-row gumbel top-k mask ----------------

__global__ void atf_select_mask(const float* __restrict__ logits,  // [B,S]
                                const int* __restrict__ k_sel_ptr,
                                float* __restrict__ mask_out) {    // [B,S]
    __shared__ float    p[S_DIM];          // 16 KB
    __shared__ unsigned key[S_DIM];        // 16 KB (order-preserving uint keys)
    __shared__ float    red[256];
    __shared__ unsigned redu[256];
    const int b = blockIdx.x, t = threadIdx.x;

    for (int s = t; s < S_DIM; s += 256) {
        float g = gumbel_from(pcg_hash(0x85EBCA6Bu ^ (unsigned)(b * S_DIM + s)));
        float v = logits[b * S_DIM + s] + g;            // perturbed (TAU = 1)
        p[s] = v;
        unsigned u = __float_as_uint(v);
        key[s] = (u & 0x80000000u) ? ~u : (u | 0x80000000u);
    }
    __syncthreads();

    // row max
    float mx = -3.4e38f;
    for (int s = t; s < S_DIM; s += 256) mx = fmaxf(mx, p[s]);
    red[t] = mx; __syncthreads();
    for (int off = 128; off > 0; off >>= 1) {
        if (t < off) red[t] = fmaxf(red[t], red[t + off]);
        __syncthreads();
    }
    mx = red[0]; __syncthreads();

    // softmax denominator
    float se = 0.0f;
    for (int s = t; s < S_DIM; s += 256) se += __expf(p[s] - mx);
    red[t] = se; __syncthreads();
    for (int off = 128; off > 0; off >>= 1) {
        if (t < off) red[t] += red[t + off];
        __syncthreads();
    }
    const float inv = 1.0f / red[0];
    __syncthreads();

    // bitwise radix-select: largest X with count(key >= X) >= k  == k-th largest
    const unsigned k = (unsigned)k_sel_ptr[0];
    unsigned X = 0u;
    for (int bit = 31; bit >= 0; --bit) {
        unsigned trial = X | (1u << bit);
        unsigned cnt = 0u;
        for (int s = t; s < S_DIM; s += 256) cnt += (key[s] >= trial) ? 1u : 0u;
        redu[t] = cnt; __syncthreads();
        for (int off = 128; off > 0; off >>= 1) {
            if (t < off) redu[t] += redu[t + off];
            __syncthreads();
        }
        if (redu[0] >= k) X = trial;
        __syncthreads();
    }

    for (int s = t; s < S_DIM; s += 256) {
        float soft = __expf(p[s] - mx) * inv;
        float hard = (key[s] >= X) ? 1.0f : 0.0f;
        mask_out[b * S_DIM + s] = (hard - soft) + soft; // sg(hard-soft)+soft
    }
}

// ---------------- kernel 5: filtered = emb * mask ----------------

__global__ void atf_apply_mask(const float* __restrict__ emb,
                               const float* __restrict__ mask,
                               float* __restrict__ out) {
    const size_t token = blockIdx.x;                    // 32768 tokens
    const float m = mask[token];
    const float4* src = reinterpret_cast<const float4*>(emb + token * D_DIM);
    float4* dst = reinterpret_cast<float4*>(out + token * D_DIM);
    const int t = threadIdx.x;                          // 256 threads * float4 = 1024
    float4 v = src[t];
    dst[t] = make_float4(v.x * m, v.y * m, v.z * m, v.w * m);
}

// ---------------- host launcher ----------------

extern "C" void kernel_launch(void* const* d_in, const int* in_sizes, int n_in,
                              void* d_out, int out_size, void* d_ws, size_t ws_size,
                              hipStream_t stream) {
    const float* emb      = (const float*)d_in[0];  // [8,4096,1024]
    const float* W1       = (const float*)d_in[1];  // [1024,2048]
    const float* b1       = (const float*)d_in[2];  // [2048]
    const float* W2       = (const float*)d_in[3];  // [2048,1]
    const float* b2       = (const float*)d_in[4];  // [1]
    const float* k_logits = (const float*)d_in[5];  // [64]
    (void)in_sizes; (void)n_in; (void)out_size;

    float* out = (float*)d_out;
    // output layout: filtered [M*D] | selection_mask [M] | expected_k [1]
    float* out_filtered = out;
    float* out_mask     = out + (size_t)M_TOTAL * D_DIM;
    float* out_ek       = out + (size_t)M_TOTAL * D_DIM + M_TOTAL;

    // workspace layout
    char* ws = (char*)d_ws; (void)ws_size;
    ushort_t* w1T   = (ushort_t*)ws;                               // 4 MB
    float*    logit = (float*)(ws + (size_t)H_DIM * D_DIM * 2);    // 128 KB
    int*      ksel  = (int*)(ws + (size_t)H_DIM * D_DIM * 2 + (size_t)M_TOTAL * 4);

    // 1) W1 -> bf16 transposed
    atf_convert_w1<<<dim3((D_DIM * H_DIM + 255) / 256), dim3(256), 0, stream>>>(W1, w1T);

    // 2) fused WMMA scorer (256 KB dynamic LDS: full-K row tile)
    const int scorerLds = ROWS_PER_BLOCK * D_DIM * (int)sizeof(ushort_t); // 262144
    hipFuncSetAttribute(reinterpret_cast<const void*>(atf_scorer),
                        hipFuncAttributeMaxDynamicSharedMemorySize, scorerLds);
    atf_scorer<<<dim3(M_TOTAL / ROWS_PER_BLOCK), dim3(SCORER_THREADS), scorerLds, stream>>>(
        emb, w1T, b1, W2, b2, logit);

    // 3) learnable-k
    atf_kselect<<<dim3(1), dim3(64), 0, stream>>>(k_logits, out_ek, ksel);

    // 4) per-row gumbel top-k mask
    atf_select_mask<<<dim3(B_DIM), dim3(256), 0, stream>>>(logit, ksel, out_mask);

    // 5) filtered embeddings
    atf_apply_mask<<<dim3(M_TOTAL), dim3(256), 0, stream>>>(emb, out_mask, out_filtered);
}
